// _MambaBlockImpl_42047729828540
// MI455X (gfx1250) — compile-verified, matrix operand order checked by reading, not run on hardware
//
#include <hip/hip_runtime.h>
#include <hip/hip_bf16.h>

typedef __attribute__((ext_vector_type(16))) _Float16 v16h;
typedef __attribute__((ext_vector_type(8)))  float    v8f;
typedef __attribute__((ext_vector_type(4)))  unsigned int u32x4;
typedef __attribute__((ext_vector_type(8)))  int      i32x8;
typedef __attribute__((ext_vector_type(4)))  int      i32x4;

#define BB  8
#define CC  64
#define DM  128
#define DI  256
#define DS  16
#define LL  16384            // H*W
#define NT  (BB * LL)        // 131072 tokens

// ---------------- WMMA fragment mapping (16x16x32 f16, wave32) -------------
// A (16x32, MxK): lane l holds row M=l&15; half i maps to
//   k = ((l>>4)&1)*8 + ((i>>3)&1)*16 + ((i>>1)&3)*2 + (i&1)
// B assumed symmetric: lane l holds col N=l&15, same k striping.
__device__ __forceinline__ int frag_k(int lane, int i) {
  int v = i >> 1, p = i & 1;
  return (((lane >> 4) & 1) << 3) + ((v >> 2) << 4) + ((v & 3) << 1) + p;
}

// row-major LDS tile gather (used only where the tile arrives via TDM)
__device__ __forceinline__ v16h load_a_lds(const _Float16* t, int ld) {
  int lane = threadIdx.x & 31;
  int row  = lane & 15;
  v16h a;
#pragma unroll
  for (int i = 0; i < 16; ++i) a[i] = t[row * ld + frag_k(lane, i)];
  return a;
}

// fragment-ordered load: 32B contiguous per lane -> 2x ds/global_load_b128
__device__ __forceinline__ v16h load_frag16(const _Float16* base) {
  return *(const v16h*)(base + (threadIdx.x & 31) * 16);
}

__device__ __forceinline__ v8f wmma_f16(v16h a, v16h b, v8f c) {
  return __builtin_amdgcn_wmma_f32_16x16x32_f16(false, a, false, b, (short)0, c,
                                                false, false);
}

__device__ __forceinline__ float silu(float v) {
  return v / (1.f + __expf(-v));
}

// swizzle helpers: element (row,kk) -> (lane, half) inside one 16x32 K-step
__device__ __forceinline__ int swz_lane(int row, int kk) {
  return (row & 15) | ((kk & 8) << 1);
}
__device__ __forceinline__ int swz_half(int kk) {
  return (kk & 7) | ((kk >> 1) & 8);
}

// ---------------- prep: weights -> fragment-ordered f16 --------------------
// out[((ntile*ksteps + kb)*32 + lane)*16 + i] = (f16) W[n*K + k]
__global__ __launch_bounds__(256) void k_prep_bfrag(
    const float* __restrict__ W, _Float16* __restrict__ out, int N, int K) {
  int idx = blockIdx.x * 256 + threadIdx.x;
  if (idx >= N * K) return;
  int i    = idx & 15;
  int lane = (idx >> 4) & 31;
  int tile = idx >> 9;
  int ksteps = K >> 5;
  int ntile = tile / ksteps, kb = tile - ntile * ksteps;
  int col = lane & 15;
  out[idx] = (_Float16)W[(ntile * 16 + col) * K + kb * 32 + frag_k(lane, i)];
}

// Wc = po_w (64x128) @ out_proj_w (128x256), emitted directly in frag order
__global__ __launch_bounds__(256) void k_prep_wc_frag(
    const float* __restrict__ out_proj_w, const float* __restrict__ po_w,
    _Float16* __restrict__ out) {
  int idx = blockIdx.x * 256 + threadIdx.x;   // 64*256 = 16384
  int i    = idx & 15;
  int lane = (idx >> 4) & 31;
  int tile = idx >> 9;                        // ntile*8 + kb
  int ntile = tile >> 3, kb = tile & 7;
  int c = ntile * 16 + (lane & 15);
  int k = kb * 32 + frag_k(lane, i);
  float acc = 0.f;
#pragma unroll 8
  for (int m = 0; m < DM; ++m) acc += po_w[c * DM + m] * out_proj_w[m * DI + k];
  out[idx] = (_Float16)acc;
}

// ---------------- kernel 1: input proj (64->128) + LayerNorm, WMMA ---------
__global__ __launch_bounds__(256) void k_proj_ln(
    const float* __restrict__ x, const _Float16* __restrict__ pw_frag,
    const float* __restrict__ proj_b, const float* __restrict__ ln_g,
    const float* __restrict__ ln_b, _Float16* __restrict__ xm16) {
  __shared__ alignas(32) _Float16 sA[2 * 512];   // 2 K-steps, frag order
  __shared__ float sXm[16 * 128];
  __shared__ float sSum[16 * 16], sSq[16 * 16];
  __shared__ float sMu[16], sRs[16];

  int tid = threadIdx.x;
  int t0  = blockIdx.x * 16;
  // gather 16 tokens x 64 channels from NCHW x, store swizzled
#pragma unroll
  for (int r = 0; r < 4; ++r) {
    int idx = r * 256 + tid;
    int tok = idx >> 6, c = idx & 63;
    int t = t0 + tok;
    int b = t / LL, hw = t % LL;
    float v = x[((size_t)b * CC + c) * LL + hw];
    int kb = c >> 5, kk = c & 31;
    sA[kb * 512 + swz_lane(tok, kk) * 16 + swz_half(kk)] = (_Float16)v;
  }
  __syncthreads();

  int wid = tid >> 5;
  int n0  = wid * 16;            // 8 waves cover 128 outputs
  v8f acc = {};
#pragma unroll
  for (int kb = 0; kb < 2; ++kb) {
    v16h a = load_frag16(sA + kb * 512);
    v16h b = load_frag16(pw_frag + (wid * 2 + kb) * 512);
    acc = wmma_f16(a, b, acc);
  }
  {
    int lane = tid & 31;
    int col = lane & 15, rbase = (lane >> 4) << 3;
#pragma unroll
    for (int j = 0; j < 8; ++j)
      sXm[(rbase + j) * 128 + n0 + col] = acc[j] + proj_b[n0 + col];
  }
  __syncthreads();
  {  // LN partials
    int tok = tid >> 4, seg = tid & 15;
    float s = 0.f, q = 0.f;
#pragma unroll
    for (int j = 0; j < 8; ++j) {
      float v = sXm[tok * 128 + seg * 8 + j];
      s += v; q += v * v;
    }
    sSum[tok * 16 + seg] = s; sSq[tok * 16 + seg] = q;
  }
  __syncthreads();
  if (tid < 16) {
    float s = 0.f, q = 0.f;
#pragma unroll
    for (int j = 0; j < 16; ++j) { s += sSum[tid * 16 + j]; q += sSq[tid * 16 + j]; }
    float mu = s * (1.f / 128.f);
    float var = q * (1.f / 128.f) - mu * mu;
    sMu[tid] = mu; sRs[tid] = rsqrtf(var + 1e-5f);
  }
  __syncthreads();
  {
    int tok = tid >> 4, seg = tid & 15;
    float mu = sMu[tok], rs = sRs[tok];
    size_t t = t0 + tok;
#pragma unroll
    for (int j = 0; j < 8; ++j) {
      int ch = seg * 8 + j;
      float v = (sXm[tok * 128 + ch] - mu) * rs * ln_g[ch] + ln_b[ch];
      xm16[t * 128 + ch] = (_Float16)v;
    }
  }
}

// ---------------- kernel 2: in_proj GEMM (128 -> 512), WMMA + TDM ----------
__global__ __launch_bounds__(256) void k_in_proj(
    const _Float16* __restrict__ xm16, const _Float16* __restrict__ w_frag,
    _Float16* __restrict__ xpart, _Float16* __restrict__ z) {
  __shared__ alignas(32) _Float16 sA[16 * 128];  // row-major, filled by TDM
  int tid = threadIdx.x;
  int t0  = blockIdx.x * 16;
  int nblk = blockIdx.y;

  if (tid < 32) {
    // Tensor Data Mover: 2D tile, 128 halves x 16 rows, f16 (data_size=2B)
    uint64_t ga = (uint64_t)(uintptr_t)(xm16 + (size_t)t0 * DM);
    uint32_t lds = (uint32_t)(uintptr_t)(&sA[0]);
    u32x4 g0;
    g0[0] = 1u;                                  // count=1, user desc
    g0[1] = lds;                                 // lds_addr
    g0[2] = (uint32_t)ga;                        // global_addr lo
    g0[3] = (uint32_t)((ga >> 32) & 0x01FFFFFFu) | (2u << 30);  // hi | type=2
    i32x8 g1;
    g1[0] = (int)(1u << 16);                     // wg_mask=0, data_size=2B
    g1[1] = (int)(128u << 16);                   // tensor_dim0 = 128 (bits 79:48)
    g1[2] = (int)(16u << 16);                    // tensor_dim1 = 16  (bits 111:80)
    g1[3] = (int)(128u << 16);                   // tile_dim0 = 128   (bits 127:112)
    g1[4] = 16;                                  // tile_dim1 = 16, tile_dim2 = 0
    g1[5] = 128;                                 // tensor_dim0_stride lo
    g1[6] = 0;
    g1[7] = 0;
    i32x4 g2 = {};
    i32x4 g3 = {};
    i32x8 g4 = {};
    __builtin_amdgcn_tensor_load_to_lds(g0, g1, g2, g3, g4, 0);
    __builtin_amdgcn_s_wait_tensorcnt(0);
  }
  __syncthreads();

  int wid = tid >> 5;
  int n = nblk * 128 + wid * 16;
  v8f acc = {};
#pragma unroll
  for (int kb = 0; kb < 4; ++kb) {
    v16h a = load_a_lds(sA + kb * 32, 128);      // row-major gather
    v16h b = load_frag16(w_frag + ((size_t)(nblk * 8 + wid) * 4 + kb) * 512);
    acc = wmma_f16(a, b, acc);
  }
  int lane = tid & 31;
  int col = lane & 15, rbase = (lane >> 4) << 3;
  int j = n + col;
#pragma unroll
  for (int r = 0; r < 8; ++r) {
    int t = t0 + rbase + r;
    int b = t / LL, hw = t % LL;
    if (j < DI) xpart[((size_t)b * DI + j) * LL + hw] = (_Float16)acc[r];
    else        z[(size_t)t * DI + (j - DI)] = (_Float16)acc[r];
  }
}

// ---------------- kernel 3: depthwise causal conv4 + SiLU ------------------
__global__ __launch_bounds__(256) void k_conv_silu(
    const _Float16* __restrict__ xpart, const float* __restrict__ conv_w,
    const float* __restrict__ conv_b, float* __restrict__ u) {
  int bd = blockIdx.x;            // B*DI = 2048
  int b = bd >> 8, d = bd & 255;
  const _Float16* row = xpart + (size_t)bd * LL;
  float w0 = conv_w[d * 4 + 0], w1 = conv_w[d * 4 + 1];
  float w2 = conv_w[d * 4 + 2], w3 = conv_w[d * 4 + 3];
  float cb = conv_b[d];
  for (int k = threadIdx.x; k < LL; k += 256) {
    float xm3 = (k >= 3) ? (float)row[k - 3] : 0.f;
    float xm2 = (k >= 2) ? (float)row[k - 2] : 0.f;
    float xm1 = (k >= 1) ? (float)row[k - 1] : 0.f;
    float x0  = (float)row[k];
    float v = w0 * xm3 + w1 * xm2 + w2 * xm1 + w3 * x0 + cb;
    u[((size_t)b * LL + k) * DI + d] = silu(v);
  }
}

// ---------------- kernel 4: x_proj (256->40) + dt_proj (8->256)+softplus ---
__global__ __launch_bounds__(64) void k_xproj_dt(
    const float* __restrict__ u, const float* __restrict__ x_proj_w,
    const float* __restrict__ dt_proj_w, const float* __restrict__ dt_proj_b,
    float* __restrict__ dt, float* __restrict__ Bm, float* __restrict__ Cm) {
  __shared__ float su[DI];
  __shared__ float sdbl[40];
  int t = blockIdx.x;
  int tid = threadIdx.x;
  const float* ur = u + (size_t)t * DI;
#pragma unroll
  for (int r = 0; r < 4; ++r) su[r * 64 + tid] = ur[r * 64 + tid];
  __syncthreads();
  if (tid < 40) {
    float acc = 0.f;
    const float* w = x_proj_w + tid * DI;
#pragma unroll 8
    for (int k = 0; k < DI; ++k) acc += su[k] * w[k];
    sdbl[tid] = acc;
  }
  __syncthreads();
#pragma unroll
  for (int r = 0; r < 4; ++r) {
    int d = r * 64 + tid;
    float acc = dt_proj_b[d];
#pragma unroll
    for (int j = 0; j < 8; ++j) acc += sdbl[j] * dt_proj_w[d * 8 + j];
    dt[(size_t)t * DI + d] = (acc > 20.f) ? acc : __logf(1.f + __expf(acc));
  }
  if (tid < 16)      Bm[(size_t)t * 16 + tid] = sdbl[8 + tid];
  else if (tid < 32) Cm[(size_t)t * 16 + (tid - 16)] = sdbl[24 + (tid - 16)];
}

// ---------------- kernel 5: chunked selective scan -------------------------
__global__ __launch_bounds__(256) void k_scan(
    const float* __restrict__ dt, const float* __restrict__ Bm,
    const float* __restrict__ Cm, const float* __restrict__ u,
    const float* __restrict__ A_log, const float* __restrict__ Dp,
    _Float16* __restrict__ y) {
  __shared__ float sP[2][8][16], sQ[2][8][16], sH[2][8][16];
  const int CL = LL / 8;           // 2048
  int tid = threadIdx.x;
  int p = tid >> 7;
  int c = (tid >> 4) & 7;
  int s = tid & 15;
  int pair = blockIdx.x * 2 + p;
  int b = pair >> 8, d = pair & 255;
  float Ads = -__expf(A_log[d * 16 + s]);
  float Dd  = Dp[d];
  size_t base = (size_t)b * LL;

  float Pa = 1.f, q = 0.f;
  for (int i = 0; i < CL; ++i) {
    size_t ti = base + c * CL + i;
    float dtv = dt[ti * DI + d];
    float uv  = u[ti * DI + d];
    float Bv  = Bm[ti * 16 + s];
    if ((i & 63) == 0 && i + 64 < CL) {
      __builtin_prefetch(&dt[(ti + 64) * DI + d], 0, 1);
      __builtin_prefetch(&Bm[(ti + 64) * 16 + s], 0, 1);
    }
    float dA = __expf(dtv * Ads);
    Pa *= dA;
    q = dA * q + dtv * Bv * uv;
  }
  sP[p][c][s] = Pa; sQ[p][c][s] = q;
  __syncthreads();
  if ((tid & 0x70) == 0) {
    float h = 0.f;
    sH[p][0][s] = 0.f;
#pragma unroll
    for (int cc = 0; cc < 7; ++cc) {
      h = sP[p][cc][s] * h + sQ[p][cc][s];
      sH[p][cc + 1][s] = h;
    }
  }
  __syncthreads();
  float h = sH[p][c][s];
  for (int i = 0; i < CL; ++i) {
    size_t ti = base + c * CL + i;
    float dtv = dt[ti * DI + d];
    float uv  = u[ti * DI + d];
    float Bv  = Bm[ti * 16 + s];
    float Cv  = Cm[ti * 16 + s];
    float dA = __expf(dtv * Ads);
    h = dA * h + dtv * Bv * uv;
    float part = h * Cv;
    part += __shfl_xor(part, 1, 16);
    part += __shfl_xor(part, 2, 16);
    part += __shfl_xor(part, 4, 16);
    part += __shfl_xor(part, 8, 16);
    if (s == 0) y[ti * DI + d] = (_Float16)(part + Dd * uv);
  }
}

// ---------------- kernel 6: gate + folded out-proj (256->64) + residual ----
__global__ __launch_bounds__(256) void k_gate_out(
    const _Float16* __restrict__ y, const _Float16* __restrict__ z,
    const _Float16* __restrict__ wc_frag, const float* __restrict__ po_b,
    const float* __restrict__ x, float* __restrict__ out) {
  __shared__ alignas(32) _Float16 sA[2 * 8 * 512];  // [mt][kb] frag order
  int tid = threadIdx.x;
  int t0  = blockIdx.x * 32;
#pragma unroll
  for (int r = 0; r < 32; ++r) {
    int idx = r * 256 + tid;
    int m = idx >> 8, k = idx & 255;
    float yv = (float)y[(size_t)t0 * 256 + idx];
    float zv = (float)z[(size_t)t0 * 256 + idx];
    float val = yv * silu(zv);
    int kb = k >> 5, kk = k & 31;
    sA[(((m >> 4) * 8 + kb) * 32 + swz_lane(m, kk)) * 16 + swz_half(kk)] =
        (_Float16)val;
  }
  __syncthreads();
  int wid = tid >> 5;
  int mt = wid >> 2, nt = wid & 3;      // 2 M-tiles x 4 N-tiles
  v8f acc = {};
#pragma unroll
  for (int kb = 0; kb < 8; ++kb) {
    v16h a = load_frag16(sA + (mt * 8 + kb) * 512);
    v16h b = load_frag16(wc_frag + (nt * 8 + kb) * 512);
    acc = wmma_f16(a, b, acc);
  }
  int lane = tid & 31;
  int col = lane & 15, rbase = (lane >> 4) << 3;
  int cch = nt * 16 + col;
  float pb = po_b[cch];
#pragma unroll
  for (int r = 0; r < 8; ++r) {
    int t = t0 + mt * 16 + rbase + r;
    int b = t / LL, hw = t % LL;
    size_t o = ((size_t)b * CC + cch) * LL + hw;
    out[o] = acc[r] + pb + x[o];
  }
}

// ---------------- launcher --------------------------------------------------
extern "C" void kernel_launch(void* const* d_in, const int* in_sizes, int n_in,
                              void* d_out, int out_size, void* d_ws, size_t ws_size,
                              hipStream_t stream) {
  const float* x         = (const float*)d_in[0];
  const float* proj_w    = (const float*)d_in[1];
  const float* proj_b    = (const float*)d_in[2];
  const float* ln_g      = (const float*)d_in[3];
  const float* ln_b      = (const float*)d_in[4];
  const float* in_proj_w = (const float*)d_in[5];
  const float* conv_w    = (const float*)d_in[6];
  const float* conv_b    = (const float*)d_in[7];
  const float* x_proj_w  = (const float*)d_in[8];
  const float* dt_proj_w = (const float*)d_in[9];
  const float* dt_proj_b = (const float*)d_in[10];
  const float* A_log     = (const float*)d_in[11];
  const float* Dp        = (const float*)d_in[12];
  const float* out_proj_w= (const float*)d_in[13];
  const float* po_w      = (const float*)d_in[14];
  const float* po_b      = (const float*)d_in[15];
  float* out = (float*)d_out;

  char* ws = (char*)d_ws;
  size_t off = 0;
  auto alloc = [&](size_t bytes) {
    char* p = ws + off;
    off += (bytes + 255) & ~(size_t)255;
    return p;
  };
  _Float16* ws_xm   = (_Float16*)alloc((size_t)NT * DM * 2);      //  33.5 MB
  _Float16* ws_xp   = (_Float16*)alloc((size_t)BB * DI * LL * 2); //  67 MB
  _Float16* ws_z    = (_Float16*)alloc((size_t)NT * DI * 2);      //  67 MB
  float*    ws_u    = (float*)   alloc((size_t)NT * DI * 4);      // 134 MB
  float*    ws_dt   = (float*)   alloc((size_t)NT * DI * 4);      // 134 MB
  float*    ws_B    = (float*)   alloc((size_t)NT * DS * 4);      // 8.4 MB
  float*    ws_C    = (float*)   alloc((size_t)NT * DS * 4);      // 8.4 MB
  _Float16* ws_y    = (_Float16*)alloc((size_t)NT * DI * 2);      //  67 MB
  _Float16* ws_wi   = (_Float16*)alloc((size_t)2 * DI * DM * 2);  // in_proj frag
  _Float16* ws_pw   = (_Float16*)alloc((size_t)DM * CC * 2);      // proj frag
  _Float16* ws_wc   = (_Float16*)alloc((size_t)CC * DI * 2);      // Wc frag

  k_prep_bfrag  <<<dim3(256), dim3(256), 0, stream>>>(in_proj_w, ws_wi, 2 * DI, DM);
  k_prep_bfrag  <<<dim3(32),  dim3(256), 0, stream>>>(proj_w, ws_pw, DM, CC);
  k_prep_wc_frag<<<dim3(64),  dim3(256), 0, stream>>>(out_proj_w, po_w, ws_wc);

  k_proj_ln <<<dim3(NT / 16), dim3(256), 0, stream>>>(x, ws_pw, proj_b, ln_g,
                                                      ln_b, ws_xm);
  k_in_proj <<<dim3(NT / 16, 4), dim3(256), 0, stream>>>(ws_xm, ws_wi, ws_xp,
                                                         ws_z);
  k_conv_silu<<<dim3(BB * DI), dim3(256), 0, stream>>>(ws_xp, conv_w, conv_b,
                                                       ws_u);
  k_xproj_dt<<<dim3(NT), dim3(64), 0, stream>>>(ws_u, x_proj_w, dt_proj_w,
                                                dt_proj_b, ws_dt, ws_B, ws_C);
  k_scan    <<<dim3(BB * DI / 2), dim3(256), 0, stream>>>(ws_dt, ws_B, ws_C,
                                                          ws_u, A_log, Dp, ws_y);
  k_gate_out<<<dim3(NT / 32), dim3(256), 0, stream>>>(ws_y, ws_z, ws_wc, po_b,
                                                      x, out);
  (void)in_sizes; (void)n_in; (void)out_size; (void)ws_size;
}
